// InteractionBlockOp_48421461295176
// MI455X (gfx1250) — matrix operand order repaired and checked by reading, not compiled, not run on hardware
//
#include <hip/hip_runtime.h>
#include <hip/hip_bf16.h>

#define HID 128
#define NG 50
#define NNODES 65536
#define NEDGES 1048576

typedef __attribute__((ext_vector_type(16))) __bf16 v16bf;
typedef __attribute__((ext_vector_type(8)))  float  v8f;

__device__ __forceinline__ float sspf(float x) {
  // shifted softplus: max(x,0) + log1p(exp(-|x|)) - ln(2)
  float t = __expf(-fabsf(x));
  return fmaxf(x, 0.0f) + __logf(1.0f + t) - 0.69314718055994531f;
}

__device__ __forceinline__ v8f wmma_bf16(v16bf a, v16bf b, v8f c) {
  return __builtin_amdgcn_wmma_f32_16x16x32_bf16(false, a, false, b, (short)0, c,
                                                 false, false);
}

// ---------------------------------------------------------------------------
// Fragment helpers (ISA 7.12.2 layouts, wave32)
//  A 16x32: lane = m + 16*half; slots 0-7 -> k = kb+half*8+0..7,
//           slots 8-15 -> k = kb+16+half*8+0..7  (two contiguous 8-chunks)
//  B 32x16: lane = n + 16*(khalf); slots 0-7 -> k = kb+khalf*8+0..7, 8-15 -> kb+16+khalf*8+0..7
// ---------------------------------------------------------------------------

// A fragment from f32 global, row-major [16][ld], K multiple of 32 (no guards).
// Vectorized: 4x b128 loads per lane.
__device__ __forceinline__ v16bf a_frag_g32(const float* __restrict__ base, int ld,
                                            int kbase) {
  int lane = threadIdx.x & 31;
  int m = lane & 15;
  int half = lane >> 4;
  const float* r0 = base + (size_t)m * ld + kbase + half * 8;
  float4 c0 = *(const float4*)(r0);
  float4 c1 = *(const float4*)(r0 + 4);
  float4 c2 = *(const float4*)(r0 + 16);
  float4 c3 = *(const float4*)(r0 + 20);
  v16bf a;
  a[0] = (__bf16)c0.x;  a[1] = (__bf16)c0.y;  a[2] = (__bf16)c0.z;  a[3] = (__bf16)c0.w;
  a[4] = (__bf16)c1.x;  a[5] = (__bf16)c1.y;  a[6] = (__bf16)c1.z;  a[7] = (__bf16)c1.w;
  a[8] = (__bf16)c2.x;  a[9] = (__bf16)c2.y;  a[10] = (__bf16)c2.z; a[11] = (__bf16)c2.w;
  a[12] = (__bf16)c3.x; a[13] = (__bf16)c3.y; a[14] = (__bf16)c3.z; a[15] = (__bf16)c3.w;
  return a;
}

// A fragment from f32 global with K zero-padding (edge_attr, K=50): scalar guarded.
__device__ __forceinline__ v16bf a_frag_gpad(const float* __restrict__ base, int ld,
                                             int kbase, int kmax) {
  int lane = threadIdx.x & 31;
  int m = lane & 15;
  int half = lane >> 4;
  v16bf a;
#pragma unroll
  for (int v = 0; v < 8; ++v) {
    int k0 = kbase + ((v & 4) << 2) + half * 8 + (v & 3) * 2;
    float f0 = (k0 < kmax) ? base[(size_t)m * ld + k0] : 0.0f;
    float f1 = (k0 + 1 < kmax) ? base[(size_t)m * ld + k0 + 1] : 0.0f;
    a[2 * v] = (__bf16)f0;
    a[2 * v + 1] = (__bf16)f1;
  }
  return a;
}

// B fragment from pre-packed bf16 buffer: one 32B vector load (2x b128).
// packed layout: [ks][ntile][lane][16]
__device__ __forceinline__ v16bf b_frag_p(const __bf16* __restrict__ p, int ks,
                                          int n) {
  int lane = threadIdx.x & 31;
  return *(const v16bf*)(p + (((size_t)ks * 8 + n) * 32 + lane) * 16);
}

// A fragment from LDS stage already in fragment order: 32B vector load.
__device__ __forceinline__ v16bf a_frag_stage(const __bf16* s) {
  int lane = threadIdx.x & 31;
  return *(const v16bf*)(s + lane * 16);
}

// Stage-write index for C element (m, ng) into A-fragment-ordered LDS
// (per wave: [4 ks][32 lanes][16 slots])
__device__ __forceinline__ int stage_idx(int wave, int m, int ng) {
  int ks = ng >> 5;
  int kk = ng & 31;
  int lhalf = (kk >> 3) & 1;
  int sl = (kk & 7) | ((kk & 16) >> 1);
  return (((wave * 4 + ks) * 32) + m + (lhalf << 4)) * 16 + sl;
}

// ------------- weight pre-pack: f32 [K][128] -> bf16 B-fragment layout -------
__global__ __launch_bounds__(256) void k_pack_b(const float* __restrict__ w,
                                                __bf16* __restrict__ p, int kmax) {
  int t = blockIdx.x * 256 + threadIdx.x;  // (ks, ntile, lane)
  int lane = t & 31;
  int n = (t >> 5) & 7;
  int ks = t >> 8;
  int col = (n << 4) + (lane & 15);
  int kb = ks * 32 + ((lane >> 4) << 3);
  v16bf b;
#pragma unroll
  for (int v = 0; v < 8; ++v) {
    int k0 = kb + ((v & 4) << 2) + (v & 3) * 2;
    float f0 = (k0 < kmax) ? w[(size_t)k0 * HID + col] : 0.0f;
    float f1 = (k0 + 1 < kmax) ? w[(size_t)(k0 + 1) * HID + col] : 0.0f;
    b[2 * v] = (__bf16)f0;
    b[2 * v + 1] = (__bf16)f1;
  }
  *(v16bf*)(p + (size_t)t * 16) = b;
}

// ---------------- kernel 1: h = x @ lin1_w  [N,128]x[128,128] ----------------
__global__ __launch_bounds__(256) void k_node_gemm(const float* __restrict__ x,
                                                   const __bf16* __restrict__ pw,
                                                   float* __restrict__ h) {
  int wave = threadIdx.x >> 5;
  int lane = threadIdx.x & 31;
  int rowbase = blockIdx.x * 128 + wave * 16;
  const float* abase = x + (size_t)rowbase * HID;

  v8f acc[8] = {};
#pragma unroll
  for (int ks = 0; ks < 4; ++ks) {
    v16bf a = a_frag_g32(abase, HID, ks * 32);
#pragma unroll
    for (int n = 0; n < 8; ++n) {
      acc[n] = wmma_bf16(a, b_frag_p(pw, ks, n), acc[n]);
    }
  }
  int half = lane >> 4, nlo = lane & 15;
#pragma unroll
  for (int n = 0; n < 8; ++n) {
#pragma unroll
    for (int r = 0; r < 8; ++r) {
      int m = (half << 3) + r;
      h[(size_t)(rowbase + m) * HID + n * 16 + nlo] = acc[n][r];
    }
  }
}

// ------------- kernel 2: fused edge filter MLP + CFConv scatter --------------
__global__ __launch_bounds__(256) void k_edge(
    const float* __restrict__ edge_attr, const int* __restrict__ srcIdx,
    const int* __restrict__ dstIdx, const float* __restrict__ edge_weight,
    const __bf16* __restrict__ pw1, const float* __restrict__ b1,
    const __bf16* __restrict__ pw2, const float* __restrict__ b2,
    const float* __restrict__ h, float* __restrict__ agg) {
  __shared__ __attribute__((aligned(32))) __bf16 stage[8 * 4 * 32 * 16];  // 32 KB
  __shared__ int sSrc[128];
  __shared__ int sDst[128];
  __shared__ float sEnv[128];

  int wave = threadIdx.x >> 5;
  int lane = threadIdx.x & 31;
  int eblock = blockIdx.x * 128;

  if (threadIdx.x < 128) {
    int e = eblock + threadIdx.x;
    int s = srcIdx[e];
    sSrc[threadIdx.x] = s;
    sDst[threadIdx.x] = dstIdx[e];
    float wt = edge_weight[e];
    sEnv[threadIdx.x] = 0.5f * (__cosf(wt * 0.52359877559829887f) + 1.0f);
    // warm the gather lines (h is L2-resident) -> global_prefetch_b8
    const char* hp = (const char*)(h + (size_t)s * HID);
    __builtin_prefetch(hp, 0, 1);
    __builtin_prefetch(hp + 256, 0, 1);
  }
  __syncthreads();

  int ebase = eblock + wave * 16;
  const float* abase = edge_attr + (size_t)ebase * NG;

  // GEMM1: [16,50(pad64)] x [50,128] -> t
  v8f acc[8] = {};
#pragma unroll
  for (int ks = 0; ks < 2; ++ks) {
    v16bf a = a_frag_gpad(abase, NG, ks * 32, NG);
#pragma unroll
    for (int n = 0; n < 8; ++n) {
      acc[n] = wmma_bf16(a, b_frag_p(pw1, ks, n), acc[n]);
    }
  }

  int half = lane >> 4, nlo = lane & 15;
#pragma unroll
  for (int n = 0; n < 8; ++n) {
    int ng = n * 16 + nlo;
    float bias = b1[ng];
#pragma unroll
    for (int r = 0; r < 8; ++r) {
      int m = (half << 3) + r;
      stage[stage_idx(wave, m, ng)] = (__bf16)sspf(acc[n][r] + bias);
    }
  }
  __syncthreads();

  // GEMM2: [16,128] x [128,128] -> W
  v8f acc2[8] = {};
#pragma unroll
  for (int ks = 0; ks < 4; ++ks) {
    v16bf a = a_frag_stage(&stage[(wave * 4 + ks) * 32 * 16]);
#pragma unroll
    for (int n = 0; n < 8; ++n) {
      acc2[n] = wmma_bf16(a, b_frag_p(pw2, ks, n), acc2[n]);
    }
  }

  // W = (t@w2 + b2) * Cenv ; msg = h[src] * W ; agg[dst] += msg
#pragma unroll
  for (int n = 0; n < 8; ++n) {
    int ng = n * 16 + nlo;
    float bias = b2[ng];
#pragma unroll
    for (int r = 0; r < 8; ++r) {
      int m = (half << 3) + r;
      int li = wave * 16 + m;
      float Wf = (acc2[n][r] + bias) * sEnv[li];
      int s = sSrc[li];
      int d = sDst[li];
      float msg = h[(size_t)s * HID + ng] * Wf;
      atomicAdd(&agg[(size_t)d * HID + ng], msg);
    }
  }
}

// ---------- kernel 3: out = ssp(agg@lin2_w + b) @ lin_w + b ------------------
__global__ __launch_bounds__(256) void k_out(const float* __restrict__ agg,
                                             const __bf16* __restrict__ pw2,
                                             const float* __restrict__ b2,
                                             const __bf16* __restrict__ pw3,
                                             const float* __restrict__ b3,
                                             float* __restrict__ out) {
  __shared__ __attribute__((aligned(32))) __bf16 stage[8 * 4 * 32 * 16];  // 32 KB
  int wave = threadIdx.x >> 5;
  int lane = threadIdx.x & 31;
  int rowbase = blockIdx.x * 128 + wave * 16;
  const float* abase = agg + (size_t)rowbase * HID;

  v8f acc[8] = {};
#pragma unroll
  for (int ks = 0; ks < 4; ++ks) {
    v16bf a = a_frag_g32(abase, HID, ks * 32);
#pragma unroll
    for (int n = 0; n < 8; ++n) {
      acc[n] = wmma_bf16(a, b_frag_p(pw2, ks, n), acc[n]);
    }
  }

  int half = lane >> 4, nlo = lane & 15;
#pragma unroll
  for (int n = 0; n < 8; ++n) {
    int ng = n * 16 + nlo;
    float bias = b2[ng];
#pragma unroll
    for (int r = 0; r < 8; ++r) {
      int m = (half << 3) + r;
      stage[stage_idx(wave, m, ng)] = (__bf16)sspf(acc[n][r] + bias);
    }
  }
  __syncthreads();

  v8f acc2[8] = {};
#pragma unroll
  for (int ks = 0; ks < 4; ++ks) {
    v16bf a = a_frag_stage(&stage[(wave * 4 + ks) * 32 * 16]);
#pragma unroll
    for (int n = 0; n < 8; ++n) {
      acc2[n] = wmma_bf16(a, b_frag_p(pw3, ks, n), acc2[n]);
    }
  }

#pragma unroll
  for (int n = 0; n < 8; ++n) {
    int ng = n * 16 + nlo;
    float bias = b3[ng];
#pragma unroll
    for (int r = 0; r < 8; ++r) {
      int m = (half << 3) + r;
      out[(size_t)(rowbase + m) * HID + ng] = acc2[n][r] + bias;
    }
  }
}

extern "C" void kernel_launch(void* const* d_in, const int* in_sizes, int n_in,
                              void* d_out, int out_size, void* d_ws, size_t ws_size,
                              hipStream_t stream) {
  const float* x = (const float*)d_in[0];
  const int* eidx = (const int*)d_in[1];  // [2, E]: row0=src, row1=dst
  const float* ew = (const float*)d_in[2];
  const float* ea = (const float*)d_in[3];
  const float* mw1 = (const float*)d_in[4];
  const float* mb1 = (const float*)d_in[5];
  const float* mw2 = (const float*)d_in[6];
  const float* mb2 = (const float*)d_in[7];
  const float* l1w = (const float*)d_in[8];
  const float* l2w = (const float*)d_in[9];
  const float* l2b = (const float*)d_in[10];
  const float* lw = (const float*)d_in[11];
  const float* lb = (const float*)d_in[12];
  float* out = (float*)d_out;

  float* h = (float*)d_ws;                // 32 MB
  float* agg = h + (size_t)NNODES * HID;  // 32 MB
  // packed bf16 weight fragments after agg
  __bf16* pw1 = (__bf16*)(agg + (size_t)NNODES * HID);  // 2 ksteps: 16 KB
  __bf16* pw2 = pw1 + 2 * 256 * 16;                     // 32 KB
  __bf16* pl1 = pw2 + 4 * 256 * 16;                     // 32 KB
  __bf16* pl2 = pl1 + 4 * 256 * 16;                     // 32 KB
  __bf16* plw = pl2 + 4 * 256 * 16;                     // 32 KB

  const int* src = eidx;
  const int* dst = eidx + NEDGES;

  hipMemsetAsync(agg, 0, (size_t)NNODES * HID * sizeof(float), stream);
  k_pack_b<<<2, 256, 0, stream>>>(mw1, pw1, NG);
  k_pack_b<<<4, 256, 0, stream>>>(mw2, pw2, HID);
  k_pack_b<<<4, 256, 0, stream>>>(l1w, pl1, HID);
  k_pack_b<<<4, 256, 0, stream>>>(l2w, pl2, HID);
  k_pack_b<<<4, 256, 0, stream>>>(lw, plw, HID);

  k_node_gemm<<<NNODES / 128, 256, 0, stream>>>(x, pl1, h);
  k_edge<<<NEDGES / 128, 256, 0, stream>>>(ea, src, dst, ew, pw1, mb1, pw2, mb2,
                                           h, agg);
  k_out<<<NNODES / 128, 256, 0, stream>>>(agg, pl2, l2b, plw, lb, out);
}
